// Head_4157528343223
// MI455X (gfx1250) — matrix operand order repaired and checked by reading, not compiled, not run on hardware
//
#include <hip/hip_runtime.h>
#include <hip/hip_bf16.h>

// Problem constants (match reference)
#define BB 8
#define TT 2048
#define CC 1024
#define HH 64

typedef __attribute__((ext_vector_type(16))) __bf16 v16bf;
typedef __attribute__((ext_vector_type(8)))  __bf16 v8bf;
typedef __attribute__((ext_vector_type(8)))  float  v8f;

union FragBF {
    v16bf v;
    v8bf  h[2];
    __bf16 b[16];
};

__device__ __forceinline__ float rmax16(float v) {
    #pragma unroll
    for (int d = 1; d < 16; d <<= 1) v = fmaxf(v, __shfl_xor(v, d, 32));
    return v;
}
__device__ __forceinline__ float rsum16(float v) {
    #pragma unroll
    for (int d = 1; d < 16; d <<= 1) v += __shfl_xor(v, d, 32);
    return v;
}

// A-fragment K-base for 16-bit A 16x32 layout (ISA 7.12.2):
// lanes 0-15: V0..V3 hold K=0..7,  V4..V7 hold K=16..23
// lanes16-31: V0..V3 hold K=8..15, V4..V7 hold K=24..31
// => per lane: two contiguous runs of 8: [half*8 .. +7] and [16+half*8 .. +7]

// ---------------------------------------------------------------------------
// Stage 1: Q/K/V = x @ W  (x fp32 -> bf16 via native cvt), out bf16 [B*T][H]
// grid.x = (B*T)/16 row tiles, grid.y = 0..2 selects Wq/Wk/Wv. 32 thr/block.
// ---------------------------------------------------------------------------
__global__ void __launch_bounds__(32)
qkv_proj_kernel(const float* __restrict__ x,
                const float* __restrict__ Wq,
                const float* __restrict__ Wk,
                const float* __restrict__ Wv,
                __bf16* __restrict__ qo,
                __bf16* __restrict__ ko,
                __bf16* __restrict__ vo) {
    const int lane = threadIdx.x;
    const int m    = lane & 15;
    const int half = lane >> 4;
    const int row0 = blockIdx.x * 16;

    const float* W;
    __bf16* outp;
    if (blockIdx.y == 0)      { W = Wq; outp = qo; }
    else if (blockIdx.y == 1) { W = Wk; outp = ko; }
    else                      { W = Wv; outp = vo; }

    v8f acc0 = {}, acc1 = {}, acc2 = {}, acc3 = {};

    for (int kc = 0; kc < CC; kc += 32) {
        // A fragment: 16x32 chunk of x rows, 128-bit loads + native bf16 cvt
        FragBF a;
        const float* xrow = x + (size_t)(row0 + m) * CC + kc;
        const float4 x0 = *(const float4*)(xrow + half * 8);
        const float4 x1 = *(const float4*)(xrow + half * 8 + 4);
        const float4 x2 = *(const float4*)(xrow + 16 + half * 8);
        const float4 x3 = *(const float4*)(xrow + 16 + half * 8 + 4);
        a.b[0] = (__bf16)x0.x; a.b[1] = (__bf16)x0.y; a.b[2] = (__bf16)x0.z; a.b[3] = (__bf16)x0.w;
        a.b[4] = (__bf16)x1.x; a.b[5] = (__bf16)x1.y; a.b[6] = (__bf16)x1.z; a.b[7] = (__bf16)x1.w;
        a.b[8]  = (__bf16)x2.x; a.b[9]  = (__bf16)x2.y; a.b[10] = (__bf16)x2.z; a.b[11] = (__bf16)x2.w;
        a.b[12] = (__bf16)x3.x; a.b[13] = (__bf16)x3.y; a.b[14] = (__bf16)x3.z; a.b[15] = (__bf16)x3.w;

        // B fragments: 4 column groups of W (32x16 each); lane-coalesced loads
        FragBF b0, b1, b2, b3;
        #pragma unroll
        for (int i = 0; i < 16; ++i) {
            const float* wrow = W + (size_t)(kc + (i >> 3) * 16 + half * 8 + (i & 7)) * HH + m;
            b0.b[i] = (__bf16)wrow[0];
            b1.b[i] = (__bf16)wrow[16];
            b2.b[i] = (__bf16)wrow[32];
            b3.b[i] = (__bf16)wrow[48];
        }
        acc0 = __builtin_amdgcn_wmma_f32_16x16x32_bf16(false, a.v, false, b0.v, (short)0, acc0, false, false);
        acc1 = __builtin_amdgcn_wmma_f32_16x16x32_bf16(false, a.v, false, b1.v, (short)0, acc1, false, false);
        acc2 = __builtin_amdgcn_wmma_f32_16x16x32_bf16(false, a.v, false, b2.v, (short)0, acc2, false, false);
        acc3 = __builtin_amdgcn_wmma_f32_16x16x32_bf16(false, a.v, false, b3.v, (short)0, acc3, false, false);
    }

    // C/D layout: lane -> col m (+16*t), VGPR v -> row v + 8*half
    #pragma unroll
    for (int v = 0; v < 8; ++v) {
        __bf16* r = outp + (size_t)(row0 + v + 8 * half) * HH + m;
        r[0]  = (__bf16)acc0[v];
        r[16] = (__bf16)acc1[v];
        r[32] = (__bf16)acc2[v];
        r[48] = (__bf16)acc3[v];
    }
}

// ---------------------------------------------------------------------------
// Stage 2: causal flash attention over bf16 Q/K/V, fp32 output.
// grid.x = T/16 query tiles, grid.y = batch. 32 threads (1 wave) per block.
// ---------------------------------------------------------------------------
__global__ void __launch_bounds__(32)
flash_attn_kernel(const __bf16* __restrict__ Q,
                  const __bf16* __restrict__ K,
                  const __bf16* __restrict__ V,
                  float* __restrict__ out) {
    __shared__ __bf16 Plds[16 * 32];  // P tile staged for A-layout reload

    const int lane = threadIdx.x;
    const int m    = lane & 15;
    const int half = lane >> 4;
    const int b    = blockIdx.y;
    const int q0   = blockIdx.x * 16;
    const float scale = 0.125f;  // 1/sqrt(64)

    const __bf16* Qb = Q + (size_t)b * TT * HH;
    const __bf16* Kb = K + (size_t)b * TT * HH;
    const __bf16* Vb = V + (size_t)b * TT * HH;

    // Q A-fragments for the two 32-wide h-chunks (128-bit loads)
    FragBF aq0, aq1;
    {
        const __bf16* qrow = Qb + (size_t)(q0 + m) * HH;
        aq0.h[0] = *(const v8bf*)(qrow + half * 16);
        aq0.h[1] = *(const v8bf*)(qrow + half * 16 + 8);
        aq1.h[0] = *(const v8bf*)(qrow + 32 + half * 16);
        aq1.h[1] = *(const v8bf*)(qrow + 32 + half * 16 + 8);
    }

    v8f acc0 = {}, acc1 = {}, acc2 = {}, acc3 = {};
    float mrow[8], lrow[8];
    #pragma unroll
    for (int v = 0; v < 8; ++v) { mrow[v] = -3.0e38f; lrow[v] = 0.0f; }

    const int ntiles = (q0 + 15) / 32 + 1;  // 32-key tiles needed (causal)
    for (int jt = 0; jt < ntiles; ++jt) {
        const int jbase = jt * 32;

        // Prefetch next tile's K/V rows while we crunch this one
        if (jt + 1 < ntiles) {
            __builtin_prefetch(Kb + (size_t)(jbase + 32 + lane) * HH, 0, 1);
            __builtin_prefetch(Vb + (size_t)(jbase + 32 + lane) * HH, 0, 1);
        }

        // ---- S = Q . K^T for 32 keys (two 16-col C fragments) ----
        // B layout: lane -> key col (m), element i -> h row (half*16 + i):
        // 16 consecutive bf16 per lane per fragment -> two 128-bit loads each.
        FragBF bk00, bk01, bk10, bk11;  // [key-group][h-chunk]
        {
            const __bf16* k0 = Kb + (size_t)(jbase + m) * HH + half * 16;
            const __bf16* k1 = Kb + (size_t)(jbase + 16 + m) * HH + half * 16;
            bk00.h[0] = *(const v8bf*)(k0);       bk00.h[1] = *(const v8bf*)(k0 + 8);
            bk01.h[0] = *(const v8bf*)(k0 + 32);  bk01.h[1] = *(const v8bf*)(k0 + 40);
            bk10.h[0] = *(const v8bf*)(k1);       bk10.h[1] = *(const v8bf*)(k1 + 8);
            bk11.h[0] = *(const v8bf*)(k1 + 32);  bk11.h[1] = *(const v8bf*)(k1 + 40);
        }
        v8f zero = {};
        v8f s0 = __builtin_amdgcn_wmma_f32_16x16x32_bf16(false, aq0.v, false, bk00.v, (short)0, zero, false, false);
        s0     = __builtin_amdgcn_wmma_f32_16x16x32_bf16(false, aq1.v, false, bk01.v, (short)0, s0,   false, false);
        v8f s1 = __builtin_amdgcn_wmma_f32_16x16x32_bf16(false, aq0.v, false, bk10.v, (short)0, zero, false, false);
        s1     = __builtin_amdgcn_wmma_f32_16x16x32_bf16(false, aq1.v, false, bk11.v, (short)0, s1,   false, false);

        // ---- scale + causal mask + online softmax ----
        const int key0 = jbase + m;
        const int key1 = jbase + 16 + m;
        float sv0[8], sv1[8], alpha[8], mnew[8];
        #pragma unroll
        for (int v = 0; v < 8; ++v) {
            const int qrow = q0 + v + 8 * half;
            float a0 = s0[v] * scale;
            float a1 = s1[v] * scale;
            if (key0 > qrow) a0 = -3.0e38f;
            if (key1 > qrow) a1 = -3.0e38f;
            sv0[v] = a0;
            sv1[v] = a1;
            const float rmax = rmax16(fmaxf(a0, a1));
            mnew[v]  = fmaxf(mrow[v], rmax);
            alpha[v] = __expf(mrow[v] - mnew[v]);
            mrow[v]  = mnew[v];
        }

        float p0[8], p1[8];
        #pragma unroll
        for (int v = 0; v < 8; ++v) {
            p0[v] = __expf(sv0[v] - mnew[v]);
            p1[v] = __expf(sv1[v] - mnew[v]);
            lrow[v] = lrow[v] * alpha[v] + rsum16(p0[v] + p1[v]);
            // rescale O accumulators
            acc0[v] *= alpha[v];
            acc1[v] *= alpha[v];
            acc2[v] *= alpha[v];
            acc3[v] *= alpha[v];
            // stage P (row-major 16x32) in LDS for A-fragment reload
            Plds[(v + 8 * half) * 32 + m]      = (__bf16)p0[v];
            Plds[(v + 8 * half) * 32 + 16 + m] = (__bf16)p1[v];
        }
        __syncthreads();  // single wave: orders DS store->load for the compiler

        // ---- reload P in A-layout (128-bit DS loads), O += P.V ----
        FragBF ap;
        ap.h[0] = *(const v8bf*)&Plds[m * 32 + half * 16];
        ap.h[1] = *(const v8bf*)&Plds[m * 32 + half * 16 + 8];

        FragBF bv0, bv1, bv2, bv3;
        #pragma unroll
        for (int i = 0; i < 16; ++i) {
            const __bf16* vrow = Vb + (size_t)(jbase + half * 16 + i) * HH + m;
            bv0.b[i] = vrow[0];
            bv1.b[i] = vrow[16];
            bv2.b[i] = vrow[32];
            bv3.b[i] = vrow[48];
        }
        acc0 = __builtin_amdgcn_wmma_f32_16x16x32_bf16(false, ap.v, false, bv0.v, (short)0, acc0, false, false);
        acc1 = __builtin_amdgcn_wmma_f32_16x16x32_bf16(false, ap.v, false, bv1.v, (short)0, acc1, false, false);
        acc2 = __builtin_amdgcn_wmma_f32_16x16x32_bf16(false, ap.v, false, bv2.v, (short)0, acc2, false, false);
        acc3 = __builtin_amdgcn_wmma_f32_16x16x32_bf16(false, ap.v, false, bv3.v, (short)0, acc3, false, false);
        __syncthreads();  // protect Plds WAR before next tile
    }

    // ---- finalize: divide by l, store fp32 ----
    #pragma unroll
    for (int v = 0; v < 8; ++v) {
        const float inv = 1.0f / lrow[v];
        const size_t r = ((size_t)b * TT + (size_t)(q0 + v + 8 * half)) * HH;
        out[r + m +  0] = acc0[v] * inv;
        out[r + m + 16] = acc1[v] * inv;
        out[r + m + 32] = acc2[v] * inv;
        out[r + m + 48] = acc3[v] * inv;
    }
}

// ---------------------------------------------------------------------------
extern "C" void kernel_launch(void* const* d_in, const int* in_sizes, int n_in,
                              void* d_out, int out_size, void* d_ws, size_t ws_size,
                              hipStream_t stream) {
    const float* x  = (const float*)d_in[0];
    const float* Wq = (const float*)d_in[1];
    const float* Wk = (const float*)d_in[2];
    const float* Wv = (const float*)d_in[3];

    __bf16* qws = (__bf16*)d_ws;                       // 2 MB
    __bf16* kws = qws + (size_t)BB * TT * HH;          // 2 MB
    __bf16* vws = kws + (size_t)BB * TT * HH;          // 2 MB

    dim3 g1((BB * TT) / 16, 3);
    qkv_proj_kernel<<<g1, 32, 0, stream>>>(x, Wq, Wk, Wv, qws, kws, vws);

    dim3 g2(TT / 16, BB);
    flash_attn_kernel<<<g2, 32, 0, stream>>>(qws, kws, vws, (float*)d_out);
}
